// MCAttModel_18184891531731
// MI455X (gfx1250) — matrix operand order repaired and checked by reading, compiled, not run on hardware
//
#include <hip/hip_runtime.h>

// ---------------------------------------------------------------------------
// Problem constants (match reference)
// ---------------------------------------------------------------------------
static constexpr int BB   = 2;
static constexpr int NT   = 192;           // N tokens
static constexpr int CC   = 4;
static constexpr int EMB  = 64;
static constexpr int HD   = 128;           // HID
static constexpr int NL   = 3;
static constexpr int NAA  = 25;
static constexpr int BN   = BB * NT;       // 384
static constexpr int KPAD = 288;           // 273 padded up to multiple of 32
static constexpr int ZSTR = 296;           // Z row stride (bf16 elems), 16B-aligned rows
static constexpr int ESTR = 136;           // E row stride
static constexpr int LDSB = (NT * ZSTR + NT * ESTR) * 2;  // dynamic LDS bytes

typedef __attribute__((ext_vector_type(8)))  float          v8f;
typedef __attribute__((ext_vector_type(8)))  unsigned short v8u;
typedef __attribute__((ext_vector_type(16))) unsigned short v16u;
typedef __attribute__((ext_vector_type(16))) __bf16         v16bf;

// ---------------------------------------------------------------------------
// Helpers
// ---------------------------------------------------------------------------
__device__ __forceinline__ unsigned short f2bf(float f) {
    unsigned int u = __float_as_uint(f);
    unsigned int r = u + 0x7FFFu + ((u >> 16) & 1u);   // round-to-nearest-even
    return (unsigned short)(r >> 16);
}
__device__ __forceinline__ float bf2f(unsigned short h) {
    return __uint_as_float(((unsigned int)h) << 16);
}
// Fast silu: v_exp + v_add + v_rcp + v_mul (avoids IEEE div expansion)
__device__ __forceinline__ float silu_f(float v) {
    return v * __builtin_amdgcn_rcpf(1.0f + __expf(-v));
}

// One 16x16 output tile, K-loop of v_wmma_f32_16x16x32_bf16.
// A: row-major (M x K), per-lane two 8-elem chunks {klo..klo+7, klo+16..klo+23}
// Bt: transposed weights, row-major (N x K), per-lane two 8-elem chunks {klo..klo+7, klo+16..klo+23}
template <int KSTEPS>
__device__ __forceinline__ v8f wmma_acc(const unsigned short* __restrict__ Arow,
                                        const unsigned short* __restrict__ Brow,
                                        int aoff, int boff) {
    v8f acc = {0.f, 0.f, 0.f, 0.f, 0.f, 0.f, 0.f, 0.f};
#pragma unroll
    for (int kk = 0; kk < KSTEPS; ++kk) {
        const int k0 = kk * 32;
        v8u a0 = *(const v8u*)(Arow + k0 + aoff);
        v8u a1 = *(const v8u*)(Arow + k0 + aoff + 16);
        v8u b0 = *(const v8u*)(Brow + k0 + boff);
        v8u b1 = *(const v8u*)(Brow + k0 + boff + 16);
        v16u ua = __builtin_shufflevector(a0, a1, 0,1,2,3,4,5,6,7,8,9,10,11,12,13,14,15);
        v16u ub = __builtin_shufflevector(b0, b1, 0,1,2,3,4,5,6,7,8,9,10,11,12,13,14,15);
        acc = __builtin_amdgcn_wmma_f32_16x16x32_bf16(
            false, __builtin_bit_cast(v16bf, ua),
            false, __builtin_bit_cast(v16bf, ub),
            (short)0, acc, false, false);
    }
    return acc;
}

// D[m,n] = silu(A(MxK) @ B(KxN)) for M=192, N=128; Bt is B^T (128 x K) in global.
template <int KSTEPS>
__device__ __forceinline__ void gemm_stage(const unsigned short* __restrict__ A, int as,
                                           const unsigned short* __restrict__ Bt, int bs,
                                           unsigned short* __restrict__ D, int dsr,
                                           int lane, int wave) {
    const int aoff = (lane & 16) ? 8 : 0;
    const int boff = (lane & 16) ? 8 : 0;
    const int l15  = lane & 15;
    for (int tile = wave; tile < 96; tile += 8) {      // 12 M-tiles x 8 N-tiles
        const int mt = tile >> 3;
        const int nt = tile & 7;
        const unsigned short* Arow = A + (mt * 16 + l15) * as;
        const unsigned short* Brow = Bt + (nt * 16 + l15) * bs;
        v8f acc = wmma_acc<KSTEPS>(Arow, Brow, aoff, boff);
        const int m0 = mt * 16 + ((lane & 16) ? 8 : 0);
        const int n  = nt * 16 + l15;
#pragma unroll
        for (int v = 0; v < 8; ++v) {
            D[(m0 + v) * dsr + n] = f2bf(silu_f(acc[v]));
        }
    }
}

// ---------------------------------------------------------------------------
// Small utility kernels
// ---------------------------------------------------------------------------
__global__ void k_zero_f(float* p, int n) {
    int i = blockIdx.x * blockDim.x + threadIdx.x;
    if (i < n) p[i] = 0.f;
}
__global__ void k_copy_f(const float* __restrict__ s, float* __restrict__ d, int n) {
    int i = blockIdx.x * blockDim.x + threadIdx.x;
    if (i < n) d[i] = s[i];
}

// Transpose weight (Ksrc x 128) -> bf16 (128 x Kpad), zero-padded K.
__global__ void k_transpose(const float* __restrict__ src, unsigned short* __restrict__ dst,
                            int Ksrc, int Kpad) {
    int idx = blockIdx.x * blockDim.x + threadIdx.x;
    int total = HD * Kpad;
    if (idx >= total) return;
    int n = idx / Kpad;
    int k = idx % Kpad;
    dst[n * Kpad + k] = (k < Ksrc) ? f2bf(src[k * HD + n]) : (unsigned short)0;
}

// Per-token: is_global flag + cumulative-max segment id.
__global__ void k_segs(const int* __restrict__ S, int* __restrict__ seg,
                       unsigned char* __restrict__ isg) {
    int idx = blockIdx.x * blockDim.x + threadIdx.x;
    if (idx >= BN) return;
    int b = idx / NT, i = idx % NT;
    int s = S[idx];
    isg[idx] = (s == 21 || s == 22 || s == 23) ? 1 : 0;
    int mx = 0;
    for (int j = 0; j <= i; ++j) {
        int sj = S[b * NT + j];
        int tk = (sj == 21) ? 1 : (sj == 22) ? 2 : (sj == 23) ? 3 : 0;
        mx = (tk > mx) ? tk : mx;
    }
    seg[idx] = mx;
}

__global__ void k_masks(const float* __restrict__ X, const int* __restrict__ seg,
                        const unsigned char* __restrict__ isg,
                        unsigned char* __restrict__ mC, unsigned char* __restrict__ mS,
                        unsigned char* __restrict__ mI) {
    int idx = blockIdx.x * blockDim.x + threadIdx.x;
    if (idx >= BB * NT * NT) return;
    int j  = idx % NT;
    int bi = idx / NT;
    int i  = bi % NT;
    int b  = bi / NT;
    int gi = isg[b * NT + i], gj = isg[b * NT + j];
    int si = seg[b * NT + i], sj = seg[b * NT + j];
    const float* cai = X + ((b * NT + i) * CC + 1) * 3;
    const float* caj = X + ((b * NT + j) * CC + 1) * 3;
    float dx = cai[0] - caj[0], dy = cai[1] - caj[1], dz = cai[2] - caj[2];
    float d2 = dx * dx + dy * dy + dz * dz;
    bool neq   = (i != j);
    bool not_g = !(gi || gj);
    bool same  = (si == sj);
    bool ctx   = neq && not_g && same && (d2 <= 64.0f);
    bool inter = neq && not_g && !same && (d2 <= 144.0f);
    bool glb   = neq && ((same && (gi || gj)) || (gi && gj));
    bool adj   = ((i - j == 1) || (j - i == 1));
    bool seqm  = adj && not_g && (si != 1);
    mC[idx] = (ctx || glb) ? 1 : 0;
    mS[idx] = seqm ? 1 : 0;
    mI[idx] = inter ? 1 : 0;
}

__global__ void k_embed(const int* __restrict__ S, const float* __restrict__ emb,
                        const float* __restrict__ Win, float* __restrict__ h) {
    int bi = blockIdx.x;
    int n  = threadIdx.x;   // 128
    int s  = S[bi];
    float acc = 0.f;
#pragma unroll 4
    for (int e = 0; e < EMB; ++e) acc += emb[s * EMB + e] * Win[e * HD + n];
    h[bi * HD + n] = acc;
}

// ---------------------------------------------------------------------------
// Main fused EGNN pair kernel: one workgroup per (b, i).
// ---------------------------------------------------------------------------
__global__ __launch_bounds__(256) void k_pair(
    const float* __restrict__ h, const float* __restrict__ x,
    const unsigned char* __restrict__ mask,
    const unsigned short* __restrict__ W1T,   // 128 x 288 bf16
    const unsigned short* __restrict__ W2T,   // 128 x 128 bf16
    const unsigned short* __restrict__ Wx1T,  // 128 x 128 bf16
    const float* __restrict__ Wx2,            // 128
    float feat,
    float* __restrict__ m_sum, float* __restrict__ x_num, float* __restrict__ cnt) {
    extern __shared__ __attribute__((aligned(16))) unsigned short lds[];
    unsigned short* Zs = lds;                 // NT x ZSTR
    unsigned short* Es = lds + NT * ZSTR;     // NT x ESTR
    __shared__ float hi_s[HD];
    __shared__ float xi_s[12];
    __shared__ float wx2_s[HD];
    __shared__ float w_s[NT];

    const int bi   = blockIdx.x;
    const int b    = bi / NT;
    const int t    = threadIdx.x;
    const int lane = t & 31;
    const int wave = t >> 5;

    if (t < HD) { hi_s[t] = h[bi * HD + t]; wx2_s[t] = Wx2[t]; }
    if (t < 12) xi_s[t] = x[bi * 12 + t];
    __syncthreads();

    // Phase 0: build z rows  [h_i | h_j | radial(4x4) | feat | pad]
    if (t < NT) {
        unsigned short* zr = Zs + t * ZSTR;
        const float* hj = h + (b * NT + t) * HD;
#pragma unroll 4
        for (int k = 0; k < HD; ++k) {
            zr[k]      = f2bf(hi_s[k]);
            zr[HD + k] = f2bf(hj[k]);
        }
        const float* xj = x + (b * NT + t) * 12;
        float df[12];
#pragma unroll
        for (int q = 0; q < 12; ++q) df[q] = xi_s[q] - xj[q];
#pragma unroll
        for (int c = 0; c < 4; ++c)
#pragma unroll
            for (int e = 0; e < 4; ++e) {
                float r = df[c * 3] * df[e * 3] + df[c * 3 + 1] * df[e * 3 + 1] +
                          df[c * 3 + 2] * df[e * 3 + 2];
                zr[256 + c * 4 + e] = f2bf(r);
            }
        zr[272] = f2bf(feat);
#pragma unroll
        for (int k = 273; k < ZSTR; ++k) zr[k] = 0;
    }
    __syncthreads();

    // Phase 1: Es = silu(Z @ We1)       (192x288)@(288x128)
    gemm_stage<KPAD / 32>(Zs, ZSTR, W1T, KPAD, Es, ESTR, lane, wave);
    __syncthreads();
    // Phase 2: Ms = silu(Es @ We2) -> Zs buffer   (192x128)@(128x128)
    gemm_stage<HD / 32>(Es, ESTR, W2T, HD, Zs, ZSTR, lane, wave);
    __syncthreads();
    // Phase 3: Fs = silu(Ms @ Wx1) -> Es buffer
    gemm_stage<HD / 32>(Zs, ZSTR, Wx1T, HD, Es, ESTR, lane, wave);
    __syncthreads();

    // Phase 4: w_j = Fs[j,:] . Wx2
    if (t < NT) {
        const unsigned short* fr = Es + t * ESTR;
        float s = 0.f;
#pragma unroll 4
        for (int n = 0; n < HD; ++n) s += bf2f(fr[n]) * wx2_s[n];
        w_s[t] = s;
    }
    __syncthreads();

    // Phase 5: masked accumulation into per-node outputs (exclusive owner -> no atomics)
    const unsigned char* mrow = mask + bi * NT;
    if (t < HD) {
        float s = 0.f;
        for (int j = 0; j < NT; ++j)
            if (mrow[j]) s += bf2f(Zs[j * ZSTR + t]);   // Ms lives in Zs
        m_sum[bi * HD + t] += s;
    } else if (t < HD + 12) {
        int q = t - HD;
        float s = 0.f;
        for (int j = 0; j < NT; ++j)
            if (mrow[j]) s += w_s[j] * (xi_s[q] - x[(b * NT + j) * 12 + q]);
        x_num[bi * 12 + q] += s;
    } else if (t == HD + 12) {
        float s = 0.f;
        for (int j = 0; j < NT; ++j) s += (float)mrow[j];
        cnt[bi] += s;
    }
}

// h += silu(concat(h, m_sum) @ Wh1) @ Wh2 ;  x += x_num / max(cnt,1)
__global__ __launch_bounds__(128) void k_update(
    const float* __restrict__ Wh1, const float* __restrict__ Wh2,
    const float* __restrict__ m_sum, const float* __restrict__ x_num,
    const float* __restrict__ cnt,
    float* __restrict__ h, float* __restrict__ x) {
    __shared__ float ccat[2 * HD];
    __shared__ float g[HD];
    const int bi = blockIdx.x;
    const int t  = threadIdx.x;   // 128
    ccat[t]      = h[bi * HD + t];
    ccat[HD + t] = m_sum[bi * HD + t];
    __syncthreads();
    float s = 0.f;
#pragma unroll 4
    for (int k = 0; k < 2 * HD; ++k) s += ccat[k] * Wh1[k * HD + t];
    g[t] = silu_f(s);
    if (t < 12) {
        float d = cnt[bi];
        if (d < 1.f) d = 1.f;
        x[bi * 12 + t] += x_num[bi * 12 + t] * __builtin_amdgcn_rcpf(d);
    }
    __syncthreads();
    float s2 = 0.f;
#pragma unroll 4
    for (int k = 0; k < HD; ++k) s2 += g[k] * Wh2[k * HD + t];
    h[bi * HD + t] = ccat[t] + s2;
}

__global__ void k_logits(const float* __restrict__ h, const float* __restrict__ Wout,
                         float* __restrict__ out) {
    int bi = blockIdx.x;
    int t  = threadIdx.x;   // 32
    if (t < NAA) {
        float s = 0.f;
#pragma unroll 4
        for (int n = 0; n < HD; ++n) s += h[bi * HD + n] * Wout[n * NAA + t];
        out[bi * NAA + t] = s;
    }
}

// ---------------------------------------------------------------------------
// Host launcher
// ---------------------------------------------------------------------------
extern "C" void kernel_launch(void* const* d_in, const int* in_sizes, int n_in,
                              void* d_out, int out_size, void* d_ws, size_t ws_size,
                              hipStream_t stream) {
    (void)in_sizes; (void)n_in; (void)out_size; (void)ws_size;
    const float* X       = (const float*)d_in[0];
    const int*   S       = (const int*)d_in[1];
    const float* emb     = (const float*)d_in[2];
    const float* Win     = (const float*)d_in[3];
    const float* ctx_We1 = (const float*)d_in[4];
    const float* ctx_We2 = (const float*)d_in[5];
    const float* ctx_Wx1 = (const float*)d_in[6];
    const float* ctx_Wx2 = (const float*)d_in[7];
    const float* ctx_Wh1 = (const float*)d_in[8];
    const float* ctx_Wh2 = (const float*)d_in[9];
    const float* int_We1 = (const float*)d_in[10];
    const float* int_We2 = (const float*)d_in[11];
    const float* int_Wx1 = (const float*)d_in[12];
    const float* int_Wx2 = (const float*)d_in[13];
    const float* int_Wh1 = (const float*)d_in[14];
    const float* int_Wh2 = (const float*)d_in[15];
    const float* Wout    = (const float*)d_in[16];

    // Workspace carve-up (256B aligned chunks)
    char* ws = (char*)d_ws;
    size_t o = 0;
    auto take = [&](size_t bytes) -> char* {
        char* p = ws + o;
        o = (o + bytes + 255) & ~(size_t)255;
        return p;
    };
    unsigned char*  mC    = (unsigned char*)take(BB * NT * NT);
    unsigned char*  mS    = (unsigned char*)take(BB * NT * NT);
    unsigned char*  mI    = (unsigned char*)take(BB * NT * NT);
    int*            segW  = (int*)take(BN * sizeof(int));
    unsigned char*  isgW  = (unsigned char*)take(BN);
    float*          hW    = (float*)take(BN * HD * sizeof(float));
    float*          xW    = (float*)take(BN * 12 * sizeof(float));
    float*          msumW = (float*)take(BN * HD * sizeof(float));
    float*          xnumW = (float*)take(BN * 12 * sizeof(float));
    float*          cntW  = (float*)take(BN * sizeof(float));
    unsigned short* W1Tw  = (unsigned short*)take(6 * HD * KPAD * sizeof(unsigned short));
    unsigned short* W2Tw  = (unsigned short*)take(6 * HD * HD * sizeof(unsigned short));
    unsigned short* Wx1Tw = (unsigned short*)take(6 * HD * HD * sizeof(unsigned short));

    // Weight prep: bf16 transposed (+K pad) copies, shared via L2 by all blocks
    for (int l = 0; l < NL; ++l) {
        k_transpose<<<(HD * KPAD + 255) / 256, 256, 0, stream>>>(
            ctx_We1 + (size_t)l * 273 * HD, W1Tw + (size_t)l * HD * KPAD, 273, KPAD);
        k_transpose<<<(HD * KPAD + 255) / 256, 256, 0, stream>>>(
            int_We1 + (size_t)l * 272 * HD, W1Tw + (size_t)(3 + l) * HD * KPAD, 272, KPAD);
        k_transpose<<<(HD * HD + 255) / 256, 256, 0, stream>>>(
            ctx_We2 + (size_t)l * HD * HD, W2Tw + (size_t)l * HD * HD, HD, HD);
        k_transpose<<<(HD * HD + 255) / 256, 256, 0, stream>>>(
            int_We2 + (size_t)l * HD * HD, W2Tw + (size_t)(3 + l) * HD * HD, HD, HD);
        k_transpose<<<(HD * HD + 255) / 256, 256, 0, stream>>>(
            ctx_Wx1 + (size_t)l * HD * HD, Wx1Tw + (size_t)l * HD * HD, HD, HD);
        k_transpose<<<(HD * HD + 255) / 256, 256, 0, stream>>>(
            int_Wx1 + (size_t)l * HD * HD, Wx1Tw + (size_t)(3 + l) * HD * HD, HD, HD);
    }

    // Masks, embedding, initial x
    k_segs<<<(BN + 255) / 256, 256, 0, stream>>>(S, segW, isgW);
    k_masks<<<(BB * NT * NT + 255) / 256, 256, 0, stream>>>(X, segW, isgW, mC, mS, mI);
    k_embed<<<BN, HD, 0, stream>>>(S, emb, Win, hW);
    k_copy_f<<<(BN * 12 + 255) / 256, 256, 0, stream>>>(X, xW, BN * 12);

    auto zero_acc = [&]() {
        k_zero_f<<<(BN * HD + 255) / 256, 256, 0, stream>>>(msumW, BN * HD);
        k_zero_f<<<(BN * 12 + 255) / 256, 256, 0, stream>>>(xnumW, BN * 12);
        k_zero_f<<<(BN + 255) / 256, 256, 0, stream>>>(cntW, BN);
    };

    for (int l = 0; l < NL; ++l) {
        const unsigned short* cW1 = W1Tw + (size_t)l * HD * KPAD;
        const unsigned short* cW2 = W2Tw + (size_t)l * HD * HD;
        const unsigned short* cWx = Wx1Tw + (size_t)l * HD * HD;
        const unsigned short* iW1 = W1Tw + (size_t)(3 + l) * HD * KPAD;
        const unsigned short* iW2 = W2Tw + (size_t)(3 + l) * HD * HD;
        const unsigned short* iWx = Wx1Tw + (size_t)(3 + l) * HD * HD;

        // --- ctx conv: pairs (ctx_mask, feat=0), (seq_mask, feat=1) ---
        zero_acc();
        k_pair<<<BN, 256, LDSB, stream>>>(hW, xW, mC, cW1, cW2, cWx,
                                          ctx_Wx2 + (size_t)l * HD, 0.0f,
                                          msumW, xnumW, cntW);
        k_pair<<<BN, 256, LDSB, stream>>>(hW, xW, mS, cW1, cW2, cWx,
                                          ctx_Wx2 + (size_t)l * HD, 1.0f,
                                          msumW, xnumW, cntW);
        k_update<<<BN, HD, 0, stream>>>(ctx_Wh1 + (size_t)l * 2 * HD * HD,
                                        ctx_Wh2 + (size_t)l * HD * HD,
                                        msumW, xnumW, cntW, hW, xW);

        // --- inter conv: single pair (inter_mask, no feat) ---
        zero_acc();
        k_pair<<<BN, 256, LDSB, stream>>>(hW, xW, mI, iW1, iW2, iWx,
                                          int_Wx2 + (size_t)l * HD, 0.0f,
                                          msumW, xnumW, cntW);
        k_update<<<BN, HD, 0, stream>>>(int_Wh1 + (size_t)l * 2 * HD * HD,
                                        int_Wh2 + (size_t)l * HD * HD,
                                        msumW, xnumW, cntW, hW, xW);
    }

    // Outputs: logits (B*N*25) then x (B*N*12), concatenated flat
    float* out = (float*)d_out;
    k_logits<<<BN, 32, 0, stream>>>(hW, Wout, out);
    k_copy_f<<<(BN * 12 + 255) / 256, 256, 0, stream>>>(xW, out + BN * NAA, BN * 12);
}
